// AttentionLayer_8778913153057
// MI455X (gfx1250) — matrix operand order repaired and checked by reading, not compiled, not run on hardware
//
#include <hip/hip_runtime.h>

typedef __bf16 bf16_t;
typedef __attribute__((ext_vector_type(8)))  __bf16    v8bf;
typedef __attribute__((ext_vector_type(16))) __bf16    v16bf;
typedef __attribute__((ext_vector_type(8)))  float     v8f;
typedef __attribute__((ext_vector_type(8)))  _Float16  v8h;

#define Hdim     1024
#define SEQ      21
#define ROWS     32      // two 16-row WMMA M tiles per workgroup
#define THREADS  256     // 8 waves of 32
#define TPW      8       // 64 N-tiles of 16 cols / 8 waves

// --- CDNA5 async global->LDS copy (ASYNCcnt-tracked, 16 B per lane/issue) ---
__device__ __forceinline__ uint32_t lds_off_u32(const void* p) {
    // flat pointers into LDS carry the LDS byte offset in addr[31:0]
    return (uint32_t)(uintptr_t)p;
}

__device__ __forceinline__ void async_stage(const float* __restrict__ gsrc,
                                            float* __restrict__ ldst,
                                            int t, int nfloats)
{
    for (int i = t * 4; i < nfloats; i += THREADS * 4) {
        uint32_t la = lds_off_u32(ldst + i);
        uint64_t ga = (uint64_t)(uintptr_t)(gsrc + i);
        asm volatile("global_load_async_to_lds_b128 %0, %1, off"
                     :: "v"(la), "v"(ga) : "memory");
    }
}

__device__ __forceinline__ void wait_async0() {
    asm volatile("s_wait_asynccnt 0x0" ::: "memory");
}

__device__ __forceinline__ v16bf mk_afrag(const bf16_t* a, int k0) {
    v8bf lo = *(const v8bf*)(a + k0);        // K = k0+half*16  .. +7
    v8bf hi = *(const v8bf*)(a + k0 + 8);    // K = k0+half*16+8.. +15
    return __builtin_shufflevector(lo, hi,
             0,1,2,3,4,5,6,7,8,9,10,11,12,13,14,15);
}

// Two 16x16 output tiles sharing the same B operand (in-register B reuse):
// halves V-stream traffic per WMMA.  64 x v_wmma_f32_16x16x32_bf16 per call.
__device__ __forceinline__ void gemm_rowtile2(const bf16_t* __restrict__ a0,
                                              const bf16_t* __restrict__ a1,
                                              const bf16_t* __restrict__ b_base,
                                              v8f& acc0, v8f& acc1)
{
#pragma unroll 4
    for (int k0 = 0; k0 < Hdim; k0 += 32) {
        __builtin_prefetch(b_base + k0 + 512, 0, 3);   // near-cache prefetch
        v16bf b  = *(const v16bf*)(b_base + k0);       // K = k0+half*16 .. +15
        v16bf av0 = mk_afrag(a0, k0);
        v16bf av1 = mk_afrag(a1, k0);
        acc0 = __builtin_amdgcn_wmma_f32_16x16x32_bf16(
                   false, av0, false, b, (short)0, acc0, false, false);
        acc1 = __builtin_amdgcn_wmma_f32_16x16x32_bf16(
                   false, av1, false, b, (short)0, acc1, false, false);
    }
}

// Convert + transpose W,V (fp32 [h][n]) -> bf16 [n][h] so the WMMA B-operand
// reads a contiguous K-run per lane.
__global__ void cvt_transpose_k(const float* __restrict__ W,
                                const float* __restrict__ V,
                                bf16_t* __restrict__ Wtb,
                                bf16_t* __restrict__ Vtb)
{
    int idx = blockIdx.x * blockDim.x + threadIdx.x;
    if (idx < Hdim * Hdim) {
        int n = idx >> 10, h = idx & (Hdim - 1);
        Wtb[idx] = (bf16_t)W[h * Hdim + n];
        Vtb[idx] = (bf16_t)V[h * Hdim + n];
    }
}

__global__ __launch_bounds__(THREADS)
void attn_fused_k(const float* __restrict__ cvec,
                  const float* __restrict__ allh,
                  const bf16_t* __restrict__ Wtb,
                  const bf16_t* __restrict__ Vtb,
                  const float* __restrict__ bias,
                  const float* __restrict__ wvec,
                  float* __restrict__ out, int B)
{
    __shared__ float  Xs[ROWS * Hdim];        // 128 KB: async fp32 landing buffer
    __shared__ bf16_t Albuf[ROWS * Hdim];     //  64 KB: bf16 A operand
    __shared__ float  sc[ROWS * SEQ];         // scores -> alpha

    const int t    = threadIdx.x;
    const int lane = t & 31;
    const int wave = t >> 5;
    const int b0   = blockIdx.x * ROWS;
    const size_t BH = (size_t)B * Hdim;

    // ---- async-stage c tile, convert to bf16 ----
    async_stage(cvec + (size_t)b0 * Hdim, Xs, t, ROWS * Hdim);
    for (int i = t; i < ROWS * SEQ; i += THREADS) sc[i] = 0.f;
    wait_async0();
    __syncthreads();
    for (int i = t; i < ROWS * Hdim; i += THREADS)
        Albuf[i] = (bf16_t)Xs[i];
    __syncthreads();

    // Xs is free again: start fetching h_0 while wc GEMM runs
    async_stage(allh + (size_t)b0 * Hdim, Xs, t, ROWS * Hdim);

    const int mrow = lane & 15;                 // A: M row, B/C: N col
    const int half = lane >> 4;
    const bf16_t* a_base0 = &Albuf[mrow * Hdim + half * 16];            // M 0..15
    const bf16_t* a_base1 = a_base0 + 16 * Hdim;                        // M 16..31

    // ---- wc = c @ W for this wave's 8 N-tiles x 2 M-tiles (fp16-packed) ----
    int   ncol[TPW];
    float wt_r[TPW];
    v8h   wch0[TPW], wch1[TPW];
#pragma unroll
    for (int tt = 0; tt < TPW; ++tt) {
        ncol[tt] = (wave * TPW + tt) * 16 + mrow;
        wt_r[tt] = wvec[ncol[tt]];
        v8f w0 = {0.f,0.f,0.f,0.f,0.f,0.f,0.f,0.f};
        v8f w1 = {0.f,0.f,0.f,0.f,0.f,0.f,0.f,0.f};
        gemm_rowtile2(a_base0, a_base1,
                      Wtb + (size_t)ncol[tt] * Hdim + half * 16, w0, w1);
#pragma unroll
        for (int j = 0; j < 8; ++j) { wch0[tt][j] = (_Float16)w0[j];
                                      wch1[tt][j] = (_Float16)w1[j]; }
    }

    // ---- per-sequence: vh GEMM + tanh + score dot, double-buffered ----
    for (int s = 0; s < SEQ; ++s) {
        wait_async0();           // Xs now holds h_s (this wave's part)
        __syncthreads();         // all waves' parts landed; Albuf reads done
        for (int i = t; i < ROWS * Hdim; i += THREADS)
            Albuf[i] = (bf16_t)Xs[i];
        __syncthreads();         // Albuf ready, Xs free
        if (s + 1 < SEQ)         // overlap next tile's HBM fetch with compute
            async_stage(allh + (size_t)(s + 1) * BH + (size_t)b0 * Hdim,
                        Xs, t, ROWS * Hdim);

        float part0[8] = {0,0,0,0,0,0,0,0};
        float part1[8] = {0,0,0,0,0,0,0,0};
#pragma unroll
        for (int tt = 0; tt < TPW; ++tt) {
            v8f acc0 = {0.f,0.f,0.f,0.f,0.f,0.f,0.f,0.f};
            v8f acc1 = {0.f,0.f,0.f,0.f,0.f,0.f,0.f,0.f};
            gemm_rowtile2(a_base0, a_base1,
                          Vtb + (size_t)ncol[tt] * Hdim + half * 16, acc0, acc1);
            float bs = bias[s * Hdim + ncol[tt]];
#pragma unroll
            for (int j = 0; j < 8; ++j) {
                float e0 = tanhf(acc0[j] + (float)wch0[tt][j] + bs);
                float e1 = tanhf(acc1[j] + (float)wch1[tt][j] + bs);
                part0[j] += e0 * wt_r[tt];      // dot with Wt over this lane's N
                part1[j] += e1 * wt_r[tt];
            }
        }
        // reduce over the 16 N-lanes of each half-wave; C row M = 8*half + j
#pragma unroll
        for (int j = 0; j < 8; ++j) {
            float v0 = part0[j], v1 = part1[j];
            v0 += __shfl_xor(v0, 1, 32);  v1 += __shfl_xor(v1, 1, 32);
            v0 += __shfl_xor(v0, 2, 32);  v1 += __shfl_xor(v1, 2, 32);
            v0 += __shfl_xor(v0, 4, 32);  v1 += __shfl_xor(v1, 4, 32);
            v0 += __shfl_xor(v0, 8, 32);  v1 += __shfl_xor(v1, 8, 32);
            if (mrow == 0) {
                atomicAdd(&sc[(half * 8 + j) * SEQ + s], v0);        // ds_add_f32
                atomicAdd(&sc[(16 + half * 8 + j) * SEQ + s], v1);
            }
        }
    }
    __syncthreads();

    // ---- softmax over S (one thread per batch row) ----
    if (t < ROWS) {
        float mx = -1e30f;
        for (int s = 0; s < SEQ; ++s) mx = fmaxf(mx, sc[t * SEQ + s]);
        float ex[SEQ], sum = 0.f;
        for (int s = 0; s < SEQ; ++s) { ex[s] = __expf(sc[t * SEQ + s] - mx); sum += ex[s]; }
        float inv = 1.f / sum;
        for (int s = 0; s < SEQ; ++s) sc[t * SEQ + s] = ex[s] * inv;
    }
    __syncthreads();

    // ---- y = sum_s alpha * h  (bandwidth pass; re-read likely L2-hot) ----
    for (int i = t; i < ROWS * Hdim; i += THREADS) {
        int m = i >> 10, h = i & (Hdim - 1);
        const float* src = allh + (size_t)(b0 + m) * Hdim + h;
        float acc = 0.f;
#pragma unroll
        for (int s = 0; s < SEQ; ++s)
            acc += sc[m * SEQ + s] * src[(size_t)s * BH];
        out[(size_t)(b0 + m) * Hdim + h] = acc;
    }
}

extern "C" void kernel_launch(void* const* d_in, const int* in_sizes, int n_in,
                              void* d_out, int out_size, void* d_ws, size_t ws_size,
                              hipStream_t stream)
{
    const float* c    = (const float*)d_in[0];  // [1,B,H]
    const float* allh = (const float*)d_in[1];  // [S,B,H]
    const float* W    = (const float*)d_in[2];  // [H,H]
    const float* V    = (const float*)d_in[3];  // [H,H]
    const float* bias = (const float*)d_in[4];  // [S,H]
    const float* Wt   = (const float*)d_in[5];  // [H,1]
    float* out = (float*)d_out;                 // [1,B,H]

    const int B = in_sizes[0] / Hdim;

    bf16_t* Wtb = (bf16_t*)d_ws;                       // 2 MB
    bf16_t* Vtb = Wtb + (size_t)Hdim * Hdim;           // 2 MB

    cvt_transpose_k<<<(Hdim * Hdim + 255) / 256, 256, 0, stream>>>(W, V, Wtb, Vtb);
    attn_fused_k<<<B / ROWS, THREADS, 0, stream>>>(c, allh, Wtb, Vtb, bias, Wt, out, B);
}
